// SparseLTSRouter_81844896792945
// MI455X (gfx1250) — compile-verified, matrix operand order checked by reading, not run on hardware
//
#include <hip/hip_runtime.h>
#include <hip/hip_bf16.h>

// ---------------------------------------------------------------------------
// Problem constants (from reference):
//   h:   (2, 2048, 1024) f32  -> 4096 tokens, d=1024
//   lts: (1, 8192, 1024) f32
//   W_router: (4,1024), b_router: (4,)
//   TOP_K=32, N_EXPERTS=4, entries_per_expert=2048
//   expert window e = lts rows [e*2048, e*2048+32)  (only 4 distinct windows)
// Outputs (concatenated): result (4096*1024 f32), expert_weights (4096*4 f32)
// ---------------------------------------------------------------------------

#define D_MODEL   1024
#define N_EXPERTS 4
#define TOP_K     32
#define TILE_M    16      // tokens per attention block
#define BLOCK_T   128     // 4 waves

#define USE_ASYNC_LDS 1   // gfx1250 global_load_async_to_lds_b128 path

typedef __attribute__((ext_vector_type(16))) __bf16        v16bf;
typedef __attribute__((ext_vector_type(8)))  float         v8f;
typedef __attribute__((ext_vector_type(4)))  float         f32x4;
typedef __attribute__((ext_vector_type(4)))  unsigned int  u32x4;
typedef __attribute__((ext_vector_type(2)))  unsigned int  u32x2;

__device__ __forceinline__ unsigned short f2bf(float f) {
    union { float f; unsigned u; } x; x.f = f;
    unsigned u = x.u;
    u += 0x7FFFu + ((u >> 16) & 1u);   // round-to-nearest-even
    return (unsigned short)(u >> 16);
}

// Load one 16-bit WMMA operand fragment: two contiguous 8-element (16B) runs.
// base must be 16B-aligned, k0/k1 multiples of 8 -> lowers to 2x ds_load_b128.
__device__ __forceinline__ v16bf load_frag(const unsigned short* base, int k0, int k1) {
    union { u32x4 q[2]; v16bf v; } f;
    f.q[0] = *(const u32x4*)(base + k0);
    f.q[1] = *(const u32x4*)(base + k1);
    return f.v;
}

#if USE_ASYNC_LDS
// 32-bit LDS byte address of a generic pointer known to be in LDS.
__device__ __forceinline__ unsigned lds_addr_of(const void* p) {
    return (unsigned)(unsigned long long)
           (__attribute__((address_space(3))) const void*)p;
}
// One 16-byte async global->LDS copy (per-lane addresses). ASYNCcnt-tracked.
__device__ __forceinline__ void async_copy_b128(unsigned lds_byte_addr,
                                                const void* gsrc) {
    asm volatile("global_load_async_to_lds_b128 %0, %1, off"
                 :: "v"(lds_byte_addr), "v"((unsigned long long)gsrc)
                 : "memory");
}
__device__ __forceinline__ void wait_async0() {
    asm volatile("s_wait_asynccnt 0" ::: "memory");
}
#endif

// ---------------------------------------------------------------------------
// Kernel 0: one-time conversion of the 4 expert windows to bf16, in both
// row-major (32x1024 per expert) and transposed (1024x32 per expert) layouts.
// Total 512 KB -> stays L2-resident for the attention kernel.
// ---------------------------------------------------------------------------
__global__ __launch_bounds__(256)
void prep_windows_kernel(const float* __restrict__ lts,
                         unsigned short* __restrict__ ws_w,    // [4][32][1024]
                         unsigned short* __restrict__ ws_wt)   // [4][1024][32]
{
    const int i = blockIdx.x * 256 + threadIdx.x;   // float4 chunk id
    const int chunks_per_e = TOP_K * D_MODEL / 4;   // 8192
    if (i >= N_EXPERTS * chunks_per_e) return;
    const int e   = i / chunks_per_e;
    const int rem = i - e * chunks_per_e;
    const int r   = (rem * 4) >> 10;                // window row 0..31
    const int c   = (rem * 4) & (D_MODEL - 1);      // column

    const f32x4 v = ((const f32x4*)(lts + (size_t)(e * 2048) * D_MODEL))[rem];
    const unsigned short b0 = f2bf(v.x), b1 = f2bf(v.y),
                         b2 = f2bf(v.z), b3 = f2bf(v.w);

    *(u32x2*)(ws_w + (size_t)e * TOP_K * D_MODEL + rem * 4) =
        (u32x2){ (unsigned)b0 | ((unsigned)b1 << 16),
                 (unsigned)b2 | ((unsigned)b3 << 16) };

    unsigned short* wt = ws_wt + (size_t)e * D_MODEL * TOP_K;
    wt[(c + 0) * TOP_K + r] = b0;
    wt[(c + 1) * TOP_K + r] = b1;
    wt[(c + 2) * TOP_K + r] = b2;
    wt[(c + 3) * TOP_K + r] = b3;
}

// ---------------------------------------------------------------------------
// Kernel 1: router. One wave per token (wave32).
// ---------------------------------------------------------------------------
__global__ __launch_bounds__(BLOCK_T)
void router_kernel(const float* __restrict__ h,
                   const float* __restrict__ Wr,
                   const float* __restrict__ br,
                   float* __restrict__ ew_out,      // (n_tokens, 4)
                   int*   __restrict__ eidx,        // (n_tokens,)
                   int n_tokens)
{
    const int lane = threadIdx.x & 31;
    const int wave = threadIdx.x >> 5;
    const int tok  = blockIdx.x * (BLOCK_T / 32) + wave;
    if (tok >= n_tokens) return;

    const float* hrow = h + (size_t)tok * D_MODEL;
    float acc0 = 0.f, acc1 = 0.f, acc2 = 0.f, acc3 = 0.f;
    for (int i = lane; i < D_MODEL; i += 32) {
        float hv = hrow[i];
        acc0 += hv * Wr[0 * D_MODEL + i];
        acc1 += hv * Wr[1 * D_MODEL + i];
        acc2 += hv * Wr[2 * D_MODEL + i];
        acc3 += hv * Wr[3 * D_MODEL + i];
    }
    #pragma unroll
    for (int off = 16; off > 0; off >>= 1) {
        acc0 += __shfl_xor(acc0, off, 32);
        acc1 += __shfl_xor(acc1, off, 32);
        acc2 += __shfl_xor(acc2, off, 32);
        acc3 += __shfl_xor(acc3, off, 32);
    }
    if (lane == 0) {
        float lg[N_EXPERTS] = { acc0 + br[0], acc1 + br[1], acc2 + br[2], acc3 + br[3] };
        float mx = lg[0]; int arg = 0;
        #pragma unroll
        for (int e = 1; e < N_EXPERTS; ++e)
            if (lg[e] > mx) { mx = lg[e]; arg = e; }
        float sum = 0.f, ex[N_EXPERTS];
        #pragma unroll
        for (int e = 0; e < N_EXPERTS; ++e) { ex[e] = __expf(lg[e] - mx); sum += ex[e]; }
        float inv = 1.f / sum;
        #pragma unroll
        for (int e = 0; e < N_EXPERTS; ++e)
            ew_out[(size_t)tok * N_EXPERTS + e] = ex[e] * inv;
        eidx[tok] = arg;
    }
}

// ---------------------------------------------------------------------------
// Kernel 2: tiled attention over the 4 expert windows using WMMA bf16.
// Windows arrive pre-converted in bf16 (row-major + transposed) and are pulled
// into LDS with async global->LDS b128 copies (ASYNCcnt).
// ---------------------------------------------------------------------------
#define SMEM_H    (TILE_M * D_MODEL)       // ushort count
#define SMEM_W    (TOP_K  * D_MODEL)       // ushort count
#define SMEM_WT   (D_MODEL * TOP_K)        // ushort count
#define SMEM_BYTES ((SMEM_H + SMEM_W + SMEM_WT) * 2 \
                    + 2 * TILE_M * TOP_K * 4 + TILE_M * TOP_K * 2 + TILE_M * 4)

__global__ __launch_bounds__(BLOCK_T)
void attn_kernel(const float* __restrict__ h,
                 const unsigned short* __restrict__ ws_w,   // [4][32][1024] bf16
                 const unsigned short* __restrict__ ws_wt,  // [4][1024][32] bf16
                 const int* __restrict__ eidx,
                 float* __restrict__ out)
{
    extern __shared__ char smem[];
    unsigned short* lds_h  = (unsigned short*)smem;
    unsigned short* lds_w  = lds_h  + SMEM_H;
    unsigned short* lds_wt = lds_w  + SMEM_W;
    float*          lds_s  = (float*)(lds_wt + SMEM_WT);          // [2][16][32]
    unsigned short* lds_a  = (unsigned short*)(lds_s + 2 * TILE_M * TOP_K);
    int*            lds_e  = (int*)(lds_a + TILE_M * TOP_K);

    const int tid  = threadIdx.x;
    const int lane = tid & 31;
    const int wave = tid >> 5;
    const int halfsel = lane >> 4;      // 0: lanes 0-15, 1: lanes 16-31
    const int mrow    = lane & 15;
    const int tile_base = blockIdx.x * TILE_M;     // token index base

    // --- stage h tile as bf16 (only per-block conversion left) ---
    {
        const f32x4* hsrc = (const f32x4*)(h + (size_t)tile_base * D_MODEL);
        for (int i = tid; i < TILE_M * D_MODEL / 4; i += BLOCK_T) {
            f32x4 v = hsrc[i];
            unsigned int lo = (unsigned)f2bf(v.x) | ((unsigned)f2bf(v.y) << 16);
            unsigned int hi = (unsigned)f2bf(v.z) | ((unsigned)f2bf(v.w) << 16);
            *(u32x2*)(lds_h + i * 4) = (u32x2){lo, hi};
        }
    }
    if (tid < TILE_M) lds_e[tid] = eidx[tile_base + tid];

    // persistent result accumulators: wave owns N-tiles [wave*16, wave*16+16)
    v8f racc[16];
    #pragma unroll
    for (int t = 0; t < 16; ++t) racc[t] = (v8f){0.f,0.f,0.f,0.f,0.f,0.f,0.f,0.f};

#if USE_ASYNC_LDS
    const unsigned lds_w_base  = lds_addr_of(lds_w);
    const unsigned lds_wt_base = lds_addr_of(lds_wt);
#endif

    __syncthreads();

    for (int e = 0; e < N_EXPERTS; ++e) {
        // --- stage window e (bf16, both layouts): pure copy, no conversion ---
        {
            const u32x4* srcw  = (const u32x4*)(ws_w  + (size_t)e * SMEM_W);
            const u32x4* srcwt = (const u32x4*)(ws_wt + (size_t)e * SMEM_WT);
            const int nchunks = SMEM_W / 8;                 // 4096 x 16B
#if USE_ASYNC_LDS
            for (int i = tid; i < nchunks; i += BLOCK_T) {
                async_copy_b128(lds_w_base  + (unsigned)i * 16u, srcw  + i);
                async_copy_b128(lds_wt_base + (unsigned)i * 16u, srcwt + i);
            }
            wait_async0();
#else
            u32x4* dw  = (u32x4*)lds_w;
            u32x4* dwt = (u32x4*)lds_wt;
            for (int i = tid; i < nchunks; i += BLOCK_T) {
                dw[i]  = srcw[i];
                dwt[i] = srcwt[i];
            }
#endif
        }
        __syncthreads();

        // --- phase B: scores(16x32) = H(16x1024) x W_e^T; all 4 waves ---
        // wave -> (n-tile = wave&1, K-half = wave>>1), 16 WMMAs, double-buffered
        {
            const int nt = wave & 1;
            const int kh = wave >> 1;
            const int nrow = nt * 16 + mrow;           // window row this lane serves
            const unsigned short* abase = lds_h + mrow * D_MODEL;
            const unsigned short* bbase = lds_w + nrow * D_MODEL;
            v8f sacc = (v8f){0.f,0.f,0.f,0.f,0.f,0.f,0.f,0.f};
            const int kbeg = kh * (D_MODEL / 2);
            int k0 = kbeg + halfsel * 8;
            v16bf Acur = load_frag(abase, k0, k0 + 16);
            v16bf Bcur = load_frag(bbase, k0, k0 + 16);
            #pragma unroll
            for (int it = 0; it < 16; ++it) {
                v16bf Anx, Bnx;
                if (it < 15) {
                    const int kn = k0 + 32;
                    Anx = load_frag(abase, kn, kn + 16);
                    Bnx = load_frag(bbase, kn, kn + 16);
                }
                sacc = __builtin_amdgcn_wmma_f32_16x16x32_bf16(
                           false, Acur, false, Bcur, (short)0, sacc, false, false);
                Acur = Anx; Bcur = Bnx; k0 += 32;
            }
            #pragma unroll
            for (int r = 0; r < 8; ++r)
                lds_s[kh * (TILE_M * TOP_K) + (r + 8 * halfsel) * TOP_K
                      + nt * 16 + mrow] = sacc[r];
        }
        __syncthreads();

        // --- softmax (scale 1/sqrt(1024) = 1/32) for tokens routed to e ---
        if (tid < TILE_M) {
            if (lds_e[tid] == e) {
                float s[TOP_K], mx = -3.0e38f;
                #pragma unroll
                for (int k = 0; k < TOP_K; ++k) {
                    s[k] = (lds_s[tid * TOP_K + k]
                          + lds_s[TILE_M * TOP_K + tid * TOP_K + k]) * 0.03125f;
                    mx = fmaxf(mx, s[k]);
                }
                float sum = 0.f;
                #pragma unroll
                for (int k = 0; k < TOP_K; ++k) { s[k] = __expf(s[k] - mx); sum += s[k]; }
                const float inv = 1.f / sum;
                #pragma unroll
                for (int k = 0; k < TOP_K; ++k)
                    lds_a[tid * TOP_K + k] = f2bf(s[k] * inv);
            } else {
                #pragma unroll
                for (int k = 0; k < TOP_K; ++k)
                    lds_a[tid * TOP_K + k] = 0;   // +0.0 bf16
            }
        }
        __syncthreads();

        // --- phase C: result += Attn_e(16x32) x W_e(32x1024); double-buffered ---
        {
            const int k0 = halfsel * 8;
            const int k1 = k0 + 16;
            v16bf A = load_frag(lds_a + mrow * TOP_K, k0, k1);
            v16bf Bcur = load_frag(lds_wt + ((wave * 16) * 16 + mrow) * TOP_K, k0, k1);
            #pragma unroll
            for (int t = 0; t < 16; ++t) {
                v16bf Bnx;
                if (t < 15) {
                    const int ncol = (wave * 16 + t + 1) * 16 + mrow;
                    Bnx = load_frag(lds_wt + ncol * TOP_K, k0, k1);
                }
                racc[t] = __builtin_amdgcn_wmma_f32_16x16x32_bf16(
                              false, A, false, Bcur, (short)0, racc[t], false, false);
                Bcur = Bnx;
            }
        }
        __syncthreads();   // before next iteration overwrites lds_w / lds_wt / lds_a
    }

    // --- write result tile ---
    #pragma unroll
    for (int t = 0; t < 16; ++t) {
        const int nt = wave * 16 + t;
        #pragma unroll
        for (int r = 0; r < 8; ++r) {
            const int tok = tile_base + r + 8 * halfsel;
            out[(size_t)tok * D_MODEL + nt * 16 + mrow] = racc[t][r];
        }
    }
}

// ---------------------------------------------------------------------------
// Launch
// ---------------------------------------------------------------------------
extern "C" void kernel_launch(void* const* d_in, const int* in_sizes, int n_in,
                              void* d_out, int out_size, void* d_ws, size_t ws_size,
                              hipStream_t stream) {
    const float* h   = (const float*)d_in[0];
    const float* lts = (const float*)d_in[1];
    const float* Wr  = (const float*)d_in[2];
    const float* br  = (const float*)d_in[3];

    const int n_tokens = in_sizes[0] / D_MODEL;          // 4096

    float* result = (float*)d_out;                        // (n_tokens, 1024)
    float* ew     = result + (size_t)n_tokens * D_MODEL;  // (n_tokens, 4)

    // workspace layout: eidx (16KB) | ws_w bf16 (256KB) | ws_wt bf16 (256KB)
    int*            eidx  = (int*)d_ws;
    unsigned short* ws_w  = (unsigned short*)((char*)d_ws + 16384);
    unsigned short* ws_wt = ws_w + (size_t)N_EXPERTS * TOP_K * D_MODEL;

    // Kernel 0: one-time bf16 window prep (both layouts)
    {
        const int total_chunks = N_EXPERTS * TOP_K * D_MODEL / 4;   // 32768
        prep_windows_kernel<<<(total_chunks + 255) / 256, 256, 0, stream>>>(
            lts, ws_w, ws_wt);
    }

    // Kernel 1: router (one wave per token)
    {
        const int tokens_per_block = BLOCK_T / 32;
        const int grid = (n_tokens + tokens_per_block - 1) / tokens_per_block;
        router_kernel<<<grid, BLOCK_T, 0, stream>>>(h, Wr, br, ew, eidx, n_tokens);
    }

    // Kernel 2: WMMA attention (16 tokens per block)
    {
        const int grid = n_tokens / TILE_M;               // 256
        attn_kernel<<<grid, BLOCK_T, SMEM_BYTES, stream>>>(
            h, ws_w, ws_wt, eidx, result);
    }
}
